// decoder_33973191311954
// MI455X (gfx1250) — matrix operand order repaired and checked by reading, not compiled, not run on hardware
//
#include <hip/hip_runtime.h>
#include <hip/hip_bf16.h>

// ---------------------------------------------------------------------------
// Problem constants (reference): B=32, Tfeat=52, H=E=512, V=20000, L=20, T=32
// ---------------------------------------------------------------------------
#define BATCH   32
#define FFRAMES 52
#define HDIM    512
#define EDIM    512
#define VOCAB   20000
#define LCAP    20
#define TFRAME  32
#define G4H     2048   // 4*H

typedef __attribute__((ext_vector_type(16))) __bf16 v16bf;
typedef __attribute__((ext_vector_type(8)))  __bf16 v8bf;
typedef __attribute__((ext_vector_type(8)))  float  v8f;
typedef __attribute__((ext_vector_type(4)))  float  vf4;

union V16U { v16bf v; v8bf h[2]; };

// ---------------------------------------------------------------------------
// WMMA fragment loaders (CDNA5 ISA 7.12.2, bf16 16x16x32)
// ---------------------------------------------------------------------------
__device__ __forceinline__ v16bf load_a_bf16(const __bf16* __restrict__ A,
                                             int lda, int m0, int k0) {
  int lane = threadIdx.x & 31;
  int row  = m0 + (lane & 15);
  int half = lane >> 4;
  const __bf16* p = A + (long)row * lda + k0 + half * 8;
  V16U u;
  u.h[0] = *(const v8bf*)(p);
  u.h[1] = *(const v8bf*)(p + 16);
  return u.v;
}

// Convert an A fragment from a per-lane f32 row pointer (half*8 pre-folded):
// elems 0..7 <- p[0..8), elems 8..15 <- p[16..24)  (k and k+16 halves).
__device__ __forceinline__ v16bf cvt_a_f32(const float* __restrict__ p) {
  vf4 f0 = *(const vf4*)(p);
  vf4 f1 = *(const vf4*)(p + 4);
  vf4 f2 = *(const vf4*)(p + 16);
  vf4 f3 = *(const vf4*)(p + 20);
  v16bf a;
#pragma unroll
  for (int j = 0; j < 4; ++j) {
    a[j]      = (__bf16)f0[j];
    a[4 + j]  = (__bf16)f1[j];
    a[8 + j]  = (__bf16)f2[j];
    a[12 + j] = (__bf16)f3[j];
  }
  return a;
}

// B 32x16 (KxN) from row-major weight W[N][K] (computes A @ W^T):
// lane col = n0+(lane&15); elems j=0..7 hold K = k0 + half*8 + j, j=8..15 hold K = k0 + 16 + half*8 + (j-8).
__device__ __forceinline__ v16bf load_b_frag(const __bf16* __restrict__ W,
                                             int ldw, int n0, int k0, int N) {
  int lane = threadIdx.x & 31;
  int n = n0 + (lane & 15);
  if (n >= N) n = N - 1;                // clamp (store is guarded)
  int half = lane >> 4;
  const __bf16* p = W + (long)n * ldw + k0 + half * 8;
  V16U u;
  u.h[0] = *(const v8bf*)(p);
  u.h[1] = *(const v8bf*)(p + 16);
  return u.v;
}

#define WMMA_BF16(A_, B_, C_) \
  __builtin_amdgcn_wmma_f32_16x16x32_bf16(false, (A_), false, (B_), (short)0, (C_), false, false)

// ---------------------------------------------------------------------------
// Main GEMM: C[M][N] (+bias). A is up to three f32 segments of 512 columns
// each (activation concat [x|h] / [e|f|h] folded into the loader; f32->bf16
// conversion in-register).  One wave computes a 16x(16*NT) tile.
//   NT=1: max wave-parallelism (small-N gate GEMMs, latency-bound chain)
//   NT=4: A-fragment reuse + deep load pipe (wide projection GEMM)
// K-loop unrolled 2x with explicit ping-pong register sets X/Y (no rotation
// movs, no WMMA WAR hazards).  Per-lane, per-segment A row pointers hoisted
// out of the loop.  REQUIRES K % 64 == 0 (holds: K in {512, 1024, 1536}).
// ---------------------------------------------------------------------------
template <int NT>
__global__ __launch_bounds__(256)
void gemm_f32a_wmma(const float* __restrict__ a0, long s0,
                    const float* __restrict__ a1, long s1,
                    const float* __restrict__ a2, long s2,
                    const __bf16* __restrict__ W, int ldw,
                    float* __restrict__ C, long ldc,
                    const float* __restrict__ bias,
                    int M, int N, int K) {
  int wave = blockIdx.x * (blockDim.x >> 5) + ((int)threadIdx.x >> 5);
  int ngroups = (N + 16 * NT - 1) / (16 * NT);
  int mtiles  = (M + 15) >> 4;
  if (wave >= ngroups * mtiles) return;   // wave-uniform: EXEC stays all-1s
  int m0 = (wave / ngroups) << 4;
  int n0 = (wave % ngroups) * (16 * NT);
  int lane = threadIdx.x & 31;
  int row  = m0 + (lane & 15);
  int half = lane >> 4;

  // hoisted per-lane per-segment A row pointers (half*8 folded in)
  const float* pA0 = a0 + (long)row * s0 + half * 8;
  const float* pA1 = a1 ? (a1 + (long)row * s1 + half * 8) : pA0;
  const float* pA2 = a2 ? (a2 + (long)row * s2 + half * 8) : pA0;

#define LOAD_A(K_)                                                         \
  ({ int seg_ = (K_) >> 9;                                                 \
     const float* p_ = (seg_ == 0) ? pA0 : (seg_ == 1) ? pA1 : pA2;        \
     cvt_a_f32(p_ + ((K_) & 511)); })

  v8f acc[NT];
#pragma unroll
  for (int j = 0; j < NT; ++j) acc[j] = (v8f){};

  // ping set X preloaded at k = 0
  v16bf aX = LOAD_A(0), aY;
  v16bf bX[NT], bY[NT];
#pragma unroll
  for (int j = 0; j < NT; ++j) bX[j] = load_b_frag(W, ldw, n0 + 16 * j, 0, N);

  for (int k0 = 0; k0 < K; k0 += 64) {
    // pong set Y at k0+32 (always valid: K % 64 == 0)
    aY = LOAD_A(k0 + 32);
#pragma unroll
    for (int j = 0; j < NT; ++j)
      bY[j] = load_b_frag(W, ldw, n0 + 16 * j, k0 + 32, N);

#pragma unroll
    for (int j = 0; j < NT; ++j) acc[j] = WMMA_BF16(aX, bX[j], acc[j]);

    if (k0 + 64 < K) {                  // refill ping set X at k0+64
      aX = LOAD_A(k0 + 64);
#pragma unroll
      for (int j = 0; j < NT; ++j)
        bX[j] = load_b_frag(W, ldw, n0 + 16 * j, k0 + 64, N);
    }

#pragma unroll
    for (int j = 0; j < NT; ++j) acc[j] = WMMA_BF16(aY, bY[j], acc[j]);
  }
#undef LOAD_A

  // epilogue: C/D layout VGPR r -> M = m0 + half*8 + r, col = lane&15
#pragma unroll
  for (int j = 0; j < NT; ++j) {
    int n = n0 + 16 * j + (lane & 15);
    if (n < N) {
      float bv = bias ? bias[n] : 0.0f;
#pragma unroll
      for (int r = 0; r < 8; ++r) {
        int m = m0 + half * 8 + r;
        C[(long)m * ldc + n] = acc[j][r] + bv;
      }
    }
  }
}

// ---------------------------------------------------------------------------
// bf16-A GEMM (used once for the time-invariant attention term S, M=1024)
// ---------------------------------------------------------------------------
__global__ __launch_bounds__(256)
void gemm_bf16_wmma(const __bf16* __restrict__ A, int lda,
                    const __bf16* __restrict__ W, int ldw,
                    float* __restrict__ C, long ldc,
                    int M, int N, int K) {
  int wave = blockIdx.x * (blockDim.x >> 5) + ((int)threadIdx.x >> 5);
  int ntiles = (N + 15) >> 4;
  int mtiles = (M + 15) >> 4;
  if (wave >= ntiles * mtiles) return;
  int m0 = (wave / ntiles) << 4;
  int n0 = (wave % ntiles) << 4;
  int lane = threadIdx.x & 31;
  int nw = n0 + (lane & 15); if (nw >= N) nw = N - 1;

  v8f acc = {};
  for (int k0 = 0; k0 < K; k0 += 32) {
    if (k0 + 32 < K)
      __builtin_prefetch(W + (long)nw * ldw + k0 + 32, 0, 2);  // global_prefetch_b8
    v16bf a = load_a_bf16(A, lda, m0, k0);
    v16bf b = load_b_frag(W, ldw, n0, k0, N);
    acc = WMMA_BF16(a, b, acc);
  }
  int n = n0 + (lane & 15);
  int half = lane >> 4;
  if (n < N) {
#pragma unroll
    for (int r = 0; r < 8; ++r)
      C[(long)(m0 + half * 8 + r) * ldc + n] = acc[r];
  }
}

// ---------------------------------------------------------------------------
// Elementwise / packing kernels (one-time setup only)
// ---------------------------------------------------------------------------
__global__ void zero_f32(float* p, int n) {
  int i = blockIdx.x * blockDim.x + threadIdx.x;
  if (i < n) p[i] = 0.0f;
}

__global__ void cvt_block(const float* __restrict__ src, long src_ld, long src_col0,
                          __bf16* __restrict__ dst, long dst_ld, long dst_col0,
                          int rows, int cols) {
  long i = (long)blockIdx.x * blockDim.x + threadIdx.x;
  if (i >= (long)rows * cols) return;
  long r = i / cols, c = i % cols;
  dst[r * dst_ld + dst_col0 + c] = (__bf16)src[r * src_ld + src_col0 + c];
}

// Xe_bf[b*T + t][h] = feature[b][t][h]  (first T frames), bf16
__global__ void pack_xe(const float* __restrict__ feat, __bf16* __restrict__ xe) {
  int i = blockIdx.x * blockDim.x + threadIdx.x;
  if (i >= BATCH * TFRAME * HDIM) return;
  int h = i & (HDIM - 1);
  int t = (i >> 9) & (TFRAME - 1);
  int b = i >> 14;
  xe[i] = (__bf16)feat[((long)b * FFRAMES + t) * HDIM + h];
}

// embedding gather (f32): embf[b][l][e] = emb[captions[b][l]][e]
__global__ void gather_emb(const float* __restrict__ emb, const int* __restrict__ caps,
                           float* __restrict__ out) {
  int i = blockIdx.x * blockDim.x + threadIdx.x;
  if (i >= BATCH * LCAP * EDIM) return;
  int e = i & (EDIM - 1);
  int l = (i >> 9) % LCAP;
  int b = i / (LCAP * EDIM);
  int tok = caps[b * LCAP + l];
  out[i] = emb[(long)tok * EDIM + e];
}

// Fused attention: scores = tanh(S[b,t,:] + P[b,:] + b1) . W2 ; softmax over t;
// context[b][h] = sum_t beta[t] * feature[b][t][h].  One block per batch row.
__global__ __launch_bounds__(256)
void attn_kernel(const float* __restrict__ S, const float* __restrict__ P,
                 const float* __restrict__ b1, const float* __restrict__ W2,
                 const float* __restrict__ feature, float* __restrict__ ctx) {
  __shared__ float sc[TFRAME];
  __shared__ float beta[TFRAME];
  int b   = blockIdx.x;
  int tid = threadIdx.x;           // 256 threads = 8 wave32s
  int tp  = tid >> 3;              // t' in [0,32)
  int sub = tid & 7;

  const float* Srow = S + ((long)b * TFRAME + tp) * HDIM;
  const float* Pb   = P + (long)b * HDIM;
  float s = 0.0f;
  for (int h = sub; h < HDIM; h += 8)
    s += tanhf(Srow[h] + Pb[h] + b1[h]) * W2[h];
  for (int off = 4; off > 0; off >>= 1) s += __shfl_xor(s, off, 32);
  if (sub == 0) sc[tp] = s;
  __syncthreads();

  if (tid < 32) {                  // wave32 does the 32-wide softmax
    float v = sc[tid];
    float m = v;
    for (int off = 16; off > 0; off >>= 1) m = fmaxf(m, __shfl_xor(m, off, 32));
    float e = __expf(v - m);
    float sum = e;
    for (int off = 16; off > 0; off >>= 1) sum += __shfl_xor(sum, off, 32);
    beta[tid] = e / sum;
  }
  __syncthreads();

  const float* fb = feature + (long)b * FFRAMES * HDIM;
  for (int h = tid; h < HDIM; h += 256) {
    float acc = 0.0f;
#pragma unroll 4
    for (int t = 0; t < TFRAME; ++t) acc += beta[t] * fb[t * HDIM + h];
    ctx[(long)b * HDIM + h] = acc;
  }
}

// LSTM pointwise update (biases added here; gates i,f,g,o in H-blocks)
__global__ void lstm_update(const float* __restrict__ g,
                            const float* __restrict__ bih, const float* __restrict__ bhh,
                            float* __restrict__ h, float* __restrict__ c) {
  int i = blockIdx.x * blockDim.x + threadIdx.x;
  if (i >= BATCH * HDIM) return;
  int b = i >> 9, j = i & (HDIM - 1);
  const float* gb = g + (long)b * G4H;
  float gi = gb[j]            + bih[j]            + bhh[j];
  float gf = gb[HDIM + j]     + bih[HDIM + j]     + bhh[HDIM + j];
  float gg = gb[2 * HDIM + j] + bih[2 * HDIM + j] + bhh[2 * HDIM + j];
  float go = gb[3 * HDIM + j] + bih[3 * HDIM + j] + bhh[3 * HDIM + j];
  float si = 1.0f / (1.0f + __expf(-gi));
  float sf = 1.0f / (1.0f + __expf(-gf));
  float so = 1.0f / (1.0f + __expf(-go));
  float c2 = sf * c[i] + si * tanhf(gg);
  c[i] = c2;
  h[i] = so * tanhf(c2);
}

// ---------------------------------------------------------------------------
// Host side
// ---------------------------------------------------------------------------
static inline int gsz(long n, int b) { return (int)((n + b - 1) / b); }

extern "C" void kernel_launch(void* const* d_in, const int* in_sizes, int n_in,
                              void* d_out, int out_size, void* d_ws, size_t ws_size,
                              hipStream_t stream) {
  (void)in_sizes; (void)n_in; (void)out_size; (void)ws_size;
  const float* feature = (const float*)d_in[0];   // (B,52,H)
  const int*   caps    = (const int*)  d_in[1];   // (B,L)
  const float* emb     = (const float*)d_in[2];   // (V,E)
  const float* Wih     = (const float*)d_in[3];   // (4H, E+H)
  const float* Whh     = (const float*)d_in[4];   // (4H, H)
  const float* bih     = (const float*)d_in[5];
  const float* bhh     = (const float*)d_in[6];
  const float* aW1     = (const float*)d_in[7];   // (H, 3H)
  const float* ab1     = (const float*)d_in[8];
  const float* aW2     = (const float*)d_in[9];   // (1, H)
  const float* outW    = (const float*)d_in[11];  // (V, H)
  const float* outb    = (const float*)d_in[12];
  float* out = (float*)d_out;                     // (B*19, V)

  // ---- carve workspace -------------------------------------------------
  char* wsp = (char*)d_ws;
  auto carve = [&](size_t bytes) -> void* {
    void* p = (void*)wsp;
    wsp += (bytes + 255) & ~(size_t)255;
    return p;
  };
  __bf16* Wcat1  = (__bf16*)carve((size_t)G4H * 1024 * 2);   // [Wih[:,E:] | Whh]
  __bf16* Wcat2  = (__bf16*)carve((size_t)G4H * 1536 * 2);   // [Wih | Whh]
  __bf16* W1h    = (__bf16*)carve((size_t)HDIM * 1024 * 2);  // attn_W1[:, :2H]
  __bf16* W1x    = (__bf16*)carve((size_t)HDIM * HDIM * 2);  // attn_W1[:, 2H:]
  __bf16* outWbf = (__bf16*)carve((size_t)VOCAB * HDIM * 2);
  __bf16* Xebf   = (__bf16*)carve((size_t)BATCH * TFRAME * HDIM * 2);
  float*  embf   = (float*) carve((size_t)BATCH * LCAP * EDIM * 4);
  float*  S      = (float*) carve((size_t)BATCH * TFRAME * HDIM * 4);
  float*  P      = (float*) carve((size_t)BATCH * HDIM * 4);
  float*  hbuf   = (float*) carve((size_t)BATCH * HDIM * 4);
  float*  cbuf   = (float*) carve((size_t)BATCH * HDIM * 4);
  float*  ctx    = (float*) carve((size_t)BATCH * HDIM * 4);
  float*  gbuf   = (float*) carve((size_t)BATCH * G4H * 4);

  const int TB = 256;

  // ---- one-time packs (bf16 weights, Xe panel, embedding gather) -------
  cvt_block<<<gsz((long)G4H * 512, TB), TB, 0, stream>>>(Wih, 1024, 512, Wcat1, 1024, 0, G4H, 512);
  cvt_block<<<gsz((long)G4H * 512, TB), TB, 0, stream>>>(Whh, 512, 0, Wcat1, 1024, 512, G4H, 512);
  cvt_block<<<gsz((long)G4H * 1024, TB), TB, 0, stream>>>(Wih, 1024, 0, Wcat2, 1536, 0, G4H, 1024);
  cvt_block<<<gsz((long)G4H * 512, TB), TB, 0, stream>>>(Whh, 512, 0, Wcat2, 1536, 1024, G4H, 512);
  cvt_block<<<gsz((long)HDIM * 1024, TB), TB, 0, stream>>>(aW1, 1536, 0, W1h, 1024, 0, HDIM, 1024);
  cvt_block<<<gsz((long)HDIM * 512, TB), TB, 0, stream>>>(aW1, 1536, 1024, W1x, 512, 0, HDIM, 512);
  cvt_block<<<gsz((long)VOCAB * HDIM, TB), TB, 0, stream>>>(outW, 512, 0, outWbf, 512, 0, VOCAB, 512);
  pack_xe<<<gsz((long)BATCH * TFRAME * HDIM, TB), TB, 0, stream>>>(feature, Xebf);
  gather_emb<<<gsz((long)BATCH * LCAP * EDIM, TB), TB, 0, stream>>>(emb, caps, embf);
  zero_f32<<<gsz(BATCH * HDIM, TB), TB, 0, stream>>>(hbuf, BATCH * HDIM);
  zero_f32<<<gsz(BATCH * HDIM, TB), TB, 0, stream>>>(cbuf, BATCH * HDIM);

  // NT=1 for narrow/latency-bound GEMMs (max waves), NT=4 for the wide
  // projection (A reuse; enough waves regardless).
  auto gemm = [&](const float* A0, long S0_, const float* A1, long S1_,
                  const float* A2, long S2_, const __bf16* W, int ldw,
                  float* C, long ldc, const float* bias, int M, int N, int K) {
    int mt = (M + 15) >> 4;
    if (N >= 8192) {
      int waves = mt * ((N + 63) >> 6);
      gemm_f32a_wmma<4><<<gsz(waves, 8), 256, 0, stream>>>(
          A0, S0_, A1, S1_, A2, S2_, W, ldw, C, ldc, bias, M, N, K);
    } else {
      int waves = mt * ((N + 15) >> 4);
      gemm_f32a_wmma<1><<<gsz(waves, 8), 256, 0, stream>>>(
          A0, S0_, A1, S1_, A2, S2_, W, ldw, C, ldc, bias, M, N, K);
    }
  };

  // ---- precompute time-invariant attention term: S = Xe @ W1x^T --------
  {
    int waves = ((BATCH * TFRAME + 15) >> 4) * ((HDIM + 15) >> 4);
    gemm_bf16_wmma<<<gsz(waves, 8), 256, 0, stream>>>(Xebf, HDIM, W1x, HDIM,
                                                      S, HDIM, BATCH * TFRAME, HDIM, HDIM);
  }

  // ---- phase 1: 32 attention + double-LSTM steps -----------------------
  for (int t = 0; t < TFRAME; ++t) {
    // P = [h|c] @ W1h^T   (concat folded into A loader)
    gemm(hbuf, HDIM, cbuf, HDIM, nullptr, 0, W1h, 1024, P, HDIM, nullptr,
         BATCH, HDIM, 1024);
    attn_kernel<<<BATCH, 256, 0, stream>>>(S, P, ab1, aW2, feature, ctx);
    // cell 1: g = [ctx|h] @ [Wih[:,E:]|Whh]^T
    gemm(ctx, HDIM, hbuf, HDIM, nullptr, 0, Wcat1, 1024, gbuf, G4H, nullptr,
         BATCH, G4H, 1024);
    lstm_update<<<gsz(BATCH * HDIM, TB), TB, 0, stream>>>(gbuf, bih, bhh, hbuf, cbuf);
    // cell 2: g = [Xe_t|h] @ [Wih[:,E:]|Whh]^T
    gemm(feature + (long)t * HDIM, (long)FFRAMES * HDIM, hbuf, HDIM, nullptr, 0,
         Wcat1, 1024, gbuf, G4H, nullptr, BATCH, G4H, 1024);
    lstm_update<<<gsz(BATCH * HDIM, TB), TB, 0, stream>>>(gbuf, bih, bhh, hbuf, cbuf);
  }

  // ---- phase 2: 19 decode steps with vocab projection ------------------
  for (int t = 0; t < LCAP - 1; ++t) {
    // g = [emb_t | feat_{T+t} | h] @ [Wih|Whh]^T  (3 segments)
    gemm(embf + (long)t * EDIM, (long)LCAP * EDIM,
         feature + (long)(TFRAME + t) * HDIM, (long)FFRAMES * HDIM,
         hbuf, HDIM, Wcat2, 1536, gbuf, G4H, nullptr, BATCH, G4H, 1536);
    lstm_update<<<gsz(BATCH * HDIM, TB), TB, 0, stream>>>(gbuf, bih, bhh, hbuf, cbuf);
    // logits row (b*19 + t): reshape folded into ldc, bias in epilogue
    gemm(hbuf, HDIM, nullptr, 0, nullptr, 0, outWbf, HDIM,
         out + (long)t * VOCAB, (long)(LCAP - 1) * VOCAB, outb,
         BATCH, VOCAB, HDIM);
  }
}
